// Generator_39883066310760
// MI455X (gfx1250) — matrix operand order repaired and checked by reading, not compiled, not run on hardware
//
#include <hip/hip_runtime.h>

typedef __bf16 v16bf __attribute__((ext_vector_type(16)));
typedef __bf16 v8bf  __attribute__((ext_vector_type(8)));
typedef float  v8f   __attribute__((ext_vector_type(8)));
typedef float  v4f   __attribute__((ext_vector_type(4)));

#define NREL   6
#define NEDGES 500000
#define DD     64
#define WPB    8              // waves per block (wave32)
#define TPB    (WPB * 32)
#define TILES  (NEDGES / 16)  // 31250 sixteen-edge tiles per relation

__device__ __forceinline__ v8f wmma_bf16(v16bf a, v16bf b, v8f c) {
  // D = A(16x32 bf16) * B(32x16 bf16) + C(16x16 f32)
  return __builtin_amdgcn_wmma_f32_16x16x32_bf16(false, a, false, b, (short)0, c,
                                                 false, false);
}

// ---- A fragment from an LDS 16x64 bf16 tile (row-major, stride 64) ----
// ISA 7.12.2 16-bit A 16x32: lanes 0-15 hold K = k0+0..7 (elems 0-7) and
// k0+16..23 (elems 8-15); lanes 16-31 hold K = k0+8..15 / k0+24..31.
__device__ __forceinline__ v16bf a_frag_lds(const __bf16* rowp, int k0, int khalf) {
  v8bf lo = *(const v8bf*)(rowp + k0 + khalf);
  v8bf hi = *(const v8bf*)(rowp + k0 + 16 + khalf);
  return __builtin_shufflevector(lo, hi, 0, 1, 2, 3, 4, 5, 6, 7,
                                         8, 9, 10, 11, 12, 13, 14, 15);
}

// ---- A fragment built directly from a gathered global f32 row ----
// Gather tables stay regular-temporal: they live in the 192 MB L2.
__device__ __forceinline__ v16bf a_frag_global(const float* rowp, int k0, int khalf) {
  const v4f* p = (const v4f*)(rowp + k0 + khalf);
  v4f x0 = p[0];
  v4f x1 = p[1];
  const v4f* q = (const v4f*)(rowp + k0 + 16 + khalf);
  v4f x2 = q[0];
  v4f x3 = q[1];
  v16bf a;
#pragma unroll
  for (int i = 0; i < 4; ++i) {
    a[i]      = (__bf16)x0[i];
    a[4 + i]  = (__bf16)x1[i];
    a[8 + i]  = (__bf16)x2[i];
    a[12 + i] = (__bf16)x3[i];
  }
  return a;
}

// ---- B fragment: weights stored N-major (transposed) [n][k] bf16 in LDS ----
// B 32x16 layout (matches A): lanes 0-15: N=n0+lane,
// K = k0..k0+7 / k0+16..23; lanes 16-31: same N, K = k0+8..15 / k0+24..31.
__device__ __forceinline__ v16bf b_frag_lds(const __bf16* Bt, int n0, int k0,
                                            int lm, int hl) {
  const __bf16* p = Bt + (n0 + lm) * DD + k0 + hl * 8;
  v8bf lo = *(const v8bf*)(p);
  v8bf hi = *(const v8bf*)(p + 16);
  return __builtin_shufflevector(lo, hi, 0, 1, 2, 3, 4, 5, 6, 7,
                                         8, 9, 10, 11, 12, 13, 14, 15);
}

__device__ __forceinline__ float leaky(float x) {
  // x>0 ? x : 0.01x  ==  max(x, 0.01x)   (single v_max_num_f32 after the mul)
  return fmaxf(x, 0.01f * x);
}

__global__ __launch_bounds__(TPB) void gan_edge_score_kernel(
    const float* __restrict__ gen_h, const float* __restrict__ dis_h,
    const float* __restrict__ M_gen, const float* __restrict__ M_dis,
    const float* __restrict__ noise, const float* __restrict__ W1,
    const float* __restrict__ b1,    const float* __restrict__ W2,
    const float* __restrict__ b2,    const int* __restrict__ src,
    float* __restrict__ out) {
  // 32 KB: Mgen^T, W1^T, W2^T, Mdis^T as bf16 [n][k]
  __shared__ __bf16 sB[4][DD * DD];
  // 16 KB: per-wave activation scratch, bf16 [m][k] (16 x 64)
  __shared__ __bf16 sA[WPB][16 * DD];

  const int r     = blockIdx.y;
  const int wave  = threadIdx.x >> 5;
  const int lane  = threadIdx.x & 31;
  const int lm    = lane & 15;
  const int hl    = lane >> 4;
  const int khalf = hl * 8;  // also equals the M offset (hl*8) for C/D tiles

  // ---- stage the four 64x64 weight matrices into LDS, bf16, transposed ----
  {
    const float* srcs[4] = { M_gen + (size_t)r * DD * DD, W1, W2,
                             M_dis + (size_t)r * DD * DD };
#pragma unroll
    for (int m = 0; m < 4; ++m) {
      const float* S = srcs[m];
      for (int i = threadIdx.x; i < DD * DD; i += TPB) {
        int k = i >> 6;
        int n = i & 63;
        sB[m][n * DD + k] = (__bf16)S[i];
      }
    }
  }
  __syncthreads();

  float b1n[4], b2n[4];
#pragma unroll
  for (int t = 0; t < 4; ++t) {
    b1n[t] = b1[t * 16 + lm];
    b2n[t] = b2[t * 16 + lm];
  }

  __bf16* aScr = &sA[wave][0];
  const __bf16* arow = aScr + lm * DD;
  const int* srcR = src + (size_t)r * NEDGES;

  for (int tile = blockIdx.x * WPB + wave; tile < TILES;
       tile += gridDim.x * WPB) {
    const int base = tile * 16;
    // src is streamed exactly once -> non-temporal (keep L2 for the tables)
    const int node = __builtin_nontemporal_load(srcR + base + lm);
    const float* ghp = gen_h + (size_t)node * DD;
    const float* dhp = dis_h + (size_t)node * DD;

    // Gathered A fragments (16 edges x 64), straight from global memory (RT).
    v16bf aG0 = a_frag_global(ghp, 0, khalf);
    v16bf aG1 = a_frag_global(ghp, 32, khalf);
    v16bf aD0 = a_frag_global(dhp, 0, khalf);
    v16bf aD1 = a_frag_global(dhp, 32, khalf);

    // ---- g0 = h_src @ Mgen + noise  (noise preloaded into the accumulator) ----
    // noise is a 768 MB single-touch stream -> non-temporal loads.
    const float* np = noise + ((size_t)r * NEDGES + base) * DD;
    v8f g[4];
#pragma unroll
    for (int t = 0; t < 4; ++t) {
      v8f acc;
#pragma unroll
      for (int v = 0; v < 8; ++v)
        acc[v] = __builtin_nontemporal_load(
            np + (v + khalf) * DD + t * 16 + lm);  // (m=v+hl*8, n=t*16+lm)
      acc = wmma_bf16(aG0, b_frag_lds(sB[0], t * 16, 0, lm, hl), acc);
      acc = wmma_bf16(aG1, b_frag_lds(sB[0], t * 16, 32, lm, hl), acc);
      g[t] = acc;
    }

    // spill g0 -> LDS scratch (acc layout -> A layout), bf16
#pragma unroll
    for (int t = 0; t < 4; ++t)
#pragma unroll
      for (int v = 0; v < 8; ++v)
        aScr[(v + khalf) * DD + t * 16 + lm] = (__bf16)g[t][v];
    asm volatile("s_wait_dscnt 0" ::: "memory");

    // ---- g1 = leaky(g0 @ W1 + b1) ----
    v16bf x0 = a_frag_lds(arow, 0, khalf);
    v16bf x1 = a_frag_lds(arow, 32, khalf);
#pragma unroll
    for (int t = 0; t < 4; ++t) {
      v8f acc;
#pragma unroll
      for (int v = 0; v < 8; ++v) acc[v] = b1n[t];
      acc = wmma_bf16(x0, b_frag_lds(sB[1], t * 16, 0, lm, hl), acc);
      acc = wmma_bf16(x1, b_frag_lds(sB[1], t * 16, 32, lm, hl), acc);
#pragma unroll
      for (int v = 0; v < 8; ++v) acc[v] = leaky(acc[v]);
      g[t] = acc;
    }

    // spill g1 -> LDS scratch
#pragma unroll
    for (int t = 0; t < 4; ++t)
#pragma unroll
      for (int v = 0; v < 8; ++v)
        aScr[(v + khalf) * DD + t * 16 + lm] = (__bf16)g[t][v];
    asm volatile("s_wait_dscnt 0" ::: "memory");

    // ---- g2 = leaky(g1 @ W2 + b2) ----
    x0 = a_frag_lds(arow, 0, khalf);
    x1 = a_frag_lds(arow, 32, khalf);
#pragma unroll
    for (int t = 0; t < 4; ++t) {
      v8f acc;
#pragma unroll
      for (int v = 0; v < 8; ++v) acc[v] = b2n[t];
      acc = wmma_bf16(x0, b_frag_lds(sB[2], t * 16, 0, lm, hl), acc);
      acc = wmma_bf16(x1, b_frag_lds(sB[2], t * 16, 32, lm, hl), acc);
#pragma unroll
      for (int v = 0; v < 8; ++v) acc[v] = leaky(acc[v]);
      g[t] = acc;
    }

    // ---- s = dh_src @ Mdis ; score = rowwise dot(s, g2) ----
    float p[8];
#pragma unroll
    for (int v = 0; v < 8; ++v) p[v] = 0.0f;
#pragma unroll
    for (int t = 0; t < 4; ++t) {
      v8f sacc;
#pragma unroll
      for (int v = 0; v < 8; ++v) sacc[v] = 0.0f;
      sacc = wmma_bf16(aD0, b_frag_lds(sB[3], t * 16, 0, lm, hl), sacc);
      sacc = wmma_bf16(aD1, b_frag_lds(sB[3], t * 16, 32, lm, hl), sacc);
#pragma unroll
      for (int v = 0; v < 8; ++v) p[v] += sacc[v] * g[t][v];
    }

    // Reduce over N: lanes 0-15 hold rows M=v, lanes 16-31 hold rows M=v+8.
#pragma unroll
    for (int v = 0; v < 8; ++v) {
      float x = p[v];
#pragma unroll
      for (int m = 1; m < 16; m <<= 1) x += __shfl_xor(x, m, 32);
      if (lm == 0)
        __builtin_nontemporal_store(x, out + (size_t)r * NEDGES + base + v + khalf);
    }
  }
}

extern "C" void kernel_launch(void* const* d_in, const int* in_sizes, int n_in,
                              void* d_out, int out_size, void* d_ws,
                              size_t ws_size, hipStream_t stream) {
  const float* gen_h = (const float*)d_in[0];
  const float* dis_h = (const float*)d_in[1];
  const float* M_gen = (const float*)d_in[2];
  const float* M_dis = (const float*)d_in[3];
  const float* noise = (const float*)d_in[4];
  const float* W1    = (const float*)d_in[5];
  const float* b1    = (const float*)d_in[6];
  const float* W2    = (const float*)d_in[7];
  const float* b2    = (const float*)d_in[8];
  const int*   src   = (const int*)d_in[9];
  float* out = (float*)d_out;

  dim3 grid(128, NREL);
  gan_edge_score_kernel<<<grid, TPB, 0, stream>>>(
      gen_h, dis_h, M_gen, M_dis, noise, W1, b1, W2, b2, src, out);
}
